// Stage_8572754723432
// MI455X (gfx1250) — compile-verified, matrix operand order checked by reading, not run on hardware
//
#include <hip/hip_runtime.h>

// ---------------------------------------------------------------------------
// CDNA5 / gfx1250 implementation of the 3D cross-scan Mamba2 block.
// Dense projections (in_proj / out_proj / fc1 / fc2 / merge_conv) run through
// a wave32 WMMA GEMM (v_wmma_f32_16x16x32_f16): K templated (constant
// addressing), float4 global loads, f16 LDS staging, double-buffered tiles,
// two WMMAs per wave per K-step (A-fragment reuse). The sequential SSM scan
// is barrier-free wave-parallel VALU code with wave32 shuffle reductions.
// ---------------------------------------------------------------------------

typedef __attribute__((ext_vector_type(16))) _Float16 v16h;
typedef __attribute__((ext_vector_type(8)))  _Float16 v8h;
typedef __attribute__((ext_vector_type(4)))  _Float16 v4h;
typedef __attribute__((ext_vector_type(8)))  float    v8f;
typedef __attribute__((ext_vector_type(4)))  float    v4f;
typedef __attribute__((ext_vector_type(2)))  float    v2f;

#define CIN      48
#define NC       96          // COUT
#define CUBE     24
#define GD       12          // downsampled cube dim
#define LVOX     1728        // 12^3
#define NSCAN    12
#define TSEQ     (NSCAN*LVOX)   // 20736
#define BB       2
#define MROWS    (BB*TSEQ)      // 41472
#define DINNER   192
#define DSTATE   64
#define NHEADS   8
#define HEADDIM  24
#define CONVDIM  320
#define DINPROJ  520
#define DCONV    4
#define EPSF     1e-5f

__device__ __forceinline__ float warp_sum(float v) {
  #pragma unroll
  for (int off = 16; off > 0; off >>= 1) v += __shfl_xor(v, off, 32);
  return v;
}

// --------------------------- downsample conv + BN ---------------------------
__global__ void k_dsconv_bn(const float* __restrict__ x, const float* __restrict__ w,
                            const float* __restrict__ bias, const float* __restrict__ bnw,
                            const float* __restrict__ bnb, const float* __restrict__ bnm,
                            const float* __restrict__ bnv, float* __restrict__ xin) {
  int idx = blockIdx.x * blockDim.x + threadIdx.x;
  if (idx >= BB * NC * LVOX) return;
  int l  = idx % LVOX;
  int co = (idx / LVOX) % NC;
  int b  = idx / (LVOX * NC);
  int k = l % GD, j = (l / GD) % GD, i = l / (GD * GD);
  int i0 = 2 * i, j0 = 2 * j, k0 = 2 * k;
  float acc = bias[co];
  for (int ci = 0; ci < CIN; ++ci) {
    const float* xp = x + (((size_t)b * CIN + ci) * CUBE + i0) * (CUBE * CUBE)
                        + (size_t)j0 * CUBE + k0;
    const float* wp = w + ((size_t)co * CIN + ci) * 8;
    acc += wp[0] * xp[0]                 + wp[1] * xp[1]
         + wp[2] * xp[CUBE]              + wp[3] * xp[CUBE + 1]
         + wp[4] * xp[CUBE * CUBE]       + wp[5] * xp[CUBE * CUBE + 1]
         + wp[6] * xp[CUBE * CUBE + CUBE]+ wp[7] * xp[CUBE * CUBE + CUBE + 1];
  }
  acc = (acc - bnm[co]) * rsqrtf(bnv[co] + EPSF) * bnw[co] + bnb[co];
  xin[idx] = acc;
}

// ------------------------------- cross scan --------------------------------
__global__ void k_cross_scan(const float* __restrict__ xin, float* __restrict__ seq) {
  int idx = blockIdx.x * blockDim.x + threadIdx.x;
  if (idx >= BB * NC * LVOX) return;
  int l = idx % LVOX;
  int c = (idx / LVOX) % NC;
  int b = idx / (LVOX * NC);
  float v = xin[idx];
  int k = l % GD, j = (l / GD) % GD, i = l / (GD * GD);
  int li  = i * 144 + j * 12 + k;
  int lc  = i * 144 + k * 12 + j;
  int lj0 = j * 144 + k * 12 + i;
  int lj1 = j * 144 + i * 12 + k;
  int lk0 = k * 144 + i * 12 + j;
  int lk1 = k * 144 + j * 12 + i;
  int ls[12] = { li, lc, LVOX - 1 - li, LVOX - 1 - lc,
                 lj0, lj1, LVOX - 1 - lj0, LVOX - 1 - lj1,
                 lk0, lk1, LVOX - 1 - lk0, LVOX - 1 - lk1 };
  #pragma unroll
  for (int s = 0; s < 12; ++s)
    seq[(((size_t)b * NSCAN + s) * LVOX + ls[s]) * NC + c] = v;
}

// ------------------------- WMMA GEMM: C = A * W^T --------------------------
// A: [M,K] f32 (M % 64 == 0), W: [N,K] f32, C: [M,N] f32. K compile-time,
// K % 32 == 0. Block: 256 threads = 8 waves; tile 64(M) x 64(N); wave (wm,wn)
// computes rows wm*16..+16, cols wn*32..+32 => 2 WMMAs / K-step with shared
// A fragment. Double-buffered f16 LDS tiles; float4 global staging loads.
#define BM 64
#define BN 64
#define BK 32

template<int K>
__global__ __launch_bounds__(256) void k_gemm_wmma(
    const float* __restrict__ A, const float* __restrict__ W,
    const float* __restrict__ bias, float* __restrict__ C,
    int N, int act) {
  __shared__ __align__(16) _Float16 As[2][BM * BK];
  __shared__ __align__(16) _Float16 Ws[2][BN * BK];
  const int tid    = threadIdx.x;
  const int lane   = tid & 31;
  const int waveId = tid >> 5;
  const int wm = waveId & 3, wn = waveId >> 2;
  const int blockM = blockIdx.x * BM;
  const int blockN = blockIdx.y * BN;

  // staging: thread moves rows ra and ra+32 (4 K-values at kk) of each tile
  const int ra = tid >> 3;           // 0..31
  const int kk = (tid & 7) << 2;     // 0,4,...,28
  const float* pA0 = A + (size_t)(blockM + ra) * K + kk;
  const float* pA1 = pA0 + (size_t)32 * K;
  const int gn0 = blockN + ra, gn1 = blockN + ra + 32;
  const float* pW0 = W + (size_t)gn0 * K + kk;
  const float* pW1 = W + (size_t)gn1 * K + kk;
  const int sOff = ra * BK + kk;

  v4f a0, a1, w0, w1;
  auto ldTile = [&](int k0) {
    a0 = *(const v4f*)(pA0 + k0);
    a1 = *(const v4f*)(pA1 + k0);
    w0 = (gn0 < N) ? *(const v4f*)(pW0 + k0) : (v4f){0.f, 0.f, 0.f, 0.f};
    w1 = (gn1 < N) ? *(const v4f*)(pW1 + k0) : (v4f){0.f, 0.f, 0.f, 0.f};
  };
  auto stTile = [&](int buf) {
    v4h h0, h1, h2, h3;
    #pragma unroll
    for (int e = 0; e < 4; ++e) {
      h0[e] = (_Float16)a0[e]; h1[e] = (_Float16)a1[e];
      h2[e] = (_Float16)w0[e]; h3[e] = (_Float16)w1[e];
    }
    *(v4h*)&As[buf][sOff]           = h0;
    *(v4h*)&As[buf][sOff + 32*BK]   = h1;
    *(v4h*)&Ws[buf][sOff]           = h2;
    *(v4h*)&Ws[buf][sOff + 32*BK]   = h3;
  };

  const int row   = lane & 15;
  const int kbsel = (lane >> 4) << 3;   // ISA 16-bit operand layout: K base 0/8
  v8f acc0 = {}, acc1 = {};

  ldTile(0);
  stTile(0);
  __syncthreads();

  constexpr int KT = K / BK;
  #pragma unroll
  for (int kt = 0; kt < KT; ++kt) {
    const int cur = kt & 1;
    if (kt + 1 < KT) ldTile((kt + 1) * BK);   // prefetch next tile into regs
    const _Float16* ap  = &As[cur][(wm * 16 + row) * BK + kbsel];
    const _Float16* bp0 = &Ws[cur][(wn * 32 + row) * BK + kbsel];
    const _Float16* bp1 = &Ws[cur][(wn * 32 + 16 + row) * BK + kbsel];
    v8h alo = *(const v8h*)ap,        ahi = *(const v8h*)(ap + 16);
    v8h b0l = *(const v8h*)bp0,       b0h = *(const v8h*)(bp0 + 16);
    v8h b1l = *(const v8h*)bp1,       b1h = *(const v8h*)(bp1 + 16);
    v16h af, bf0, bf1;
    #pragma unroll
    for (int e = 0; e < 8; ++e) {
      af[e]  = alo[e]; af[e + 8]  = ahi[e];
      bf0[e] = b0l[e]; bf0[e + 8] = b0h[e];
      bf1[e] = b1l[e]; bf1[e + 8] = b1h[e];
    }
    acc0 = __builtin_amdgcn_wmma_f32_16x16x32_f16(
        false, af, false, bf0, (short)0, acc0, false, false);
    acc1 = __builtin_amdgcn_wmma_f32_16x16x32_f16(
        false, af, false, bf1, (short)0, acc1, false, false);
    if (kt + 1 < KT) stTile(1 - cur);
    __syncthreads();
  }

  const int m0  = blockM + wm * 16 + ((lane >> 4) << 3);
  const int gna = blockN + wn * 32 + (lane & 15);
  const int gnb = gna + 16;
  #pragma unroll
  for (int half = 0; half < 2; ++half) {
    int gn = half ? gnb : gna;
    if (gn >= N) continue;
    float bv = bias ? bias[gn] : 0.f;
    v8f acc = half ? acc1 : acc0;
    #pragma unroll
    for (int v = 0; v < 8; ++v) {
      float val = acc[v] + bv;
      if (act == 1)  // exact GELU
        val = 0.5f * val * (1.f + erff(val * 0.70710678118654752f));
      C[(size_t)(m0 + v) * N + gn] = val;
    }
  }
}

// ----------------------- residual add + LayerNorm (C=96) -------------------
__global__ __launch_bounds__(256) void k_ln_residual(
    const float* __restrict__ hid, const float* res_in, float* res_out,
    const float* __restrict__ w, const float* __restrict__ b,
    float* __restrict__ out) {
  int row  = blockIdx.x * 8 + (threadIdx.x >> 5);
  int lane = threadIdx.x & 31;
  float v[3];
  #pragma unroll
  for (int t = 0; t < 3; ++t) {
    int c = lane + 32 * t;
    v[t] = hid[(size_t)row * NC + c];
    if (res_in) v[t] += res_in[(size_t)row * NC + c];
  }
  float mu = warp_sum(v[0] + v[1] + v[2]) * (1.f / NC);
  float q = 0.f;
  #pragma unroll
  for (int t = 0; t < 3; ++t) { float d = v[t] - mu; q += d * d; }
  float inv = rsqrtf(warp_sum(q) * (1.f / NC) + EPSF);
  #pragma unroll
  for (int t = 0; t < 3; ++t) {
    int c = lane + 32 * t;
    res_out[(size_t)row * NC + c] = v[t];
    out[(size_t)row * NC + c] = (v[t] - mu) * inv * w[c] + b[c];
  }
}

// --------------------- depthwise causal conv1d + SiLU ----------------------
__global__ void k_conv1d_silu(const float* __restrict__ zx, const float* __restrict__ w,
                              const float* __restrict__ bias, float* __restrict__ out) {
  int idx = blockIdx.x * blockDim.x + threadIdx.x;
  if (idx >= BB * TSEQ * CONVDIM) return;
  int ch = idx % CONVDIM;
  int t  = (idx / CONVDIM) % TSEQ;
  int b  = idx / (CONVDIM * TSEQ);
  const float* wp = w + ch * DCONV;
  float acc = bias[ch];
  #pragma unroll
  for (int q = 0; q < DCONV; ++q) {
    int tt = t - (DCONV - 1) + q;
    if (tt >= 0)
      acc += wp[q] * zx[((size_t)b * TSEQ + tt) * DINPROJ + DINNER + ch];
  }
  out[idx] = acc / (1.f + expf(-acc));   // silu
}

// ----------------------------- dt / dA precompute --------------------------
__global__ void k_dt(const float* __restrict__ zx, const float* __restrict__ dt_bias,
                     const float* __restrict__ A_log, float* __restrict__ dtb,
                     float* __restrict__ dab) {
  int idx = blockIdx.x * blockDim.x + threadIdx.x;
  if (idx >= MROWS * NHEADS) return;
  int h = idx % NHEADS;
  int r = idx / NHEADS;
  float raw = zx[(size_t)r * DINPROJ + DINNER + CONVDIM + h] + dt_bias[h];
  float dt  = (raw > 20.f) ? raw : log1pf(expf(raw));  // softplus
  dtb[idx] = dt;
  dab[idx] = expf(-expf(A_log[h]) * dt);
}

// --------------------------- sequential SSM scan ---------------------------
// One block per (batch, head); 8 independent waves, barrier-free. Wave w owns
// p in {w, w+8, w+16}; lane owns state columns n = 2*lane, 2*lane+1 (float2
// loads). y via wave32 shuffle reduction. No LDS, no barriers.
__global__ __launch_bounds__(256) void k_scan(const float* __restrict__ xbcc,
                                              const float* __restrict__ dtb,
                                              const float* __restrict__ dab,
                                              float* __restrict__ ys) {
  int b  = blockIdx.x >> 3;
  int hd = blockIdx.x & 7;
  int lane = threadIdx.x & 31;
  int wave = threadIdx.x >> 5;
  float h0[3] = {0.f, 0.f, 0.f}, h1[3] = {0.f, 0.f, 0.f};
  for (int t = 0; t < TSEQ; ++t) {
    size_t r = (size_t)b * TSEQ + t;
    const float* xr = xbcc + r * CONVDIM;
    v2f Bv = *(const v2f*)(xr + DINNER + 2 * lane);
    v2f Cv = *(const v2f*)(xr + DINNER + DSTATE + 2 * lane);
    float dtv = dtb[r * NHEADS + hd];
    float dav = dab[r * NHEADS + hd];
    #pragma unroll
    for (int jj = 0; jj < 3; ++jj) {
      int p = wave + 8 * jj;
      float xp = xr[hd * HEADDIM + p] * dtv;
      h0[jj] = h0[jj] * dav + xp * Bv[0];
      h1[jj] = h1[jj] * dav + xp * Bv[1];
      float part = warp_sum(h0[jj] * Cv[0] + h1[jj] * Cv[1]);
      if (lane == 0) ys[r * DINNER + hd * HEADDIM + p] = part;
    }
  }
}

// ----------------- y = ys + D*x; gate by silu(z); RMSNorm ------------------
__global__ __launch_bounds__(256) void k_gate_rms(
    const float* __restrict__ ys, const float* __restrict__ xbcc,
    const float* __restrict__ zx, const float* __restrict__ Dp,
    const float* __restrict__ nw, float* __restrict__ out) {
  int row  = blockIdx.x * 8 + (threadIdx.x >> 5);
  int lane = threadIdx.x & 31;
  float g[6]; float q = 0.f;
  #pragma unroll
  for (int t = 0; t < 6; ++t) {
    int c = lane + 32 * t;
    float y = ys[(size_t)row * DINNER + c]
            + Dp[c / HEADDIM] * xbcc[(size_t)row * CONVDIM + c];
    float z = zx[(size_t)row * DINPROJ + c];
    float gv = y * (z / (1.f + expf(-z)));
    g[t] = gv; q += gv * gv;
  }
  float inv = rsqrtf(warp_sum(q) * (1.f / DINNER) + EPSF);
  #pragma unroll
  for (int t = 0; t < 6; ++t) {
    int c = lane + 32 * t;
    out[(size_t)row * DINNER + c] = g[t] * inv * nw[c];
  }
}

// ------------------------------- cross merge -------------------------------
__global__ void k_cross_merge(const float* __restrict__ hid, float* __restrict__ merged) {
  int idx = blockIdx.x * blockDim.x + threadIdx.x;
  if (idx >= BB * NSCAN * LVOX * NC) return;
  int c   = idx % NC;
  int l   = (idx / NC) % LVOX;
  int blk = (idx / (NC * LVOX)) % NSCAN;
  int b   = idx / (NC * LVOX * NSCAN);
  int k = l % GD, j = (l / GD) % GD, i = l / (GD * GD);
  int li  = i * 144 + j * 12 + k;
  int lc  = i * 144 + k * 12 + j;
  int lj0 = j * 144 + k * 12 + i;
  int lj1 = j * 144 + i * 12 + k;
  int lk0 = k * 144 + i * 12 + j;
  int lk1 = k * 144 + j * 12 + i;
  int s, lsrc;
  switch (blk) {
    case 0:  s = 0;  lsrc = li;             break;
    case 1:  s = 2;  lsrc = LVOX - 1 - li;  break;
    case 2:  s = 1;  lsrc = lc;             break;
    case 3:  s = 3;  lsrc = LVOX - 1 - lc;  break;
    case 4:  s = 4;  lsrc = lj0;            break;
    case 5:  s = 6;  lsrc = LVOX - 1 - lj0; break;
    case 6:  s = 5;  lsrc = lj1;            break;
    case 7:  s = 7;  lsrc = LVOX - 1 - lj1; break;
    case 8:  s = 8;  lsrc = lk0;            break;
    case 9:  s = 10; lsrc = LVOX - 1 - lk0; break;
    case 10: s = 9;  lsrc = lk1;            break;
    default: s = 11; lsrc = LVOX - 1 - lk1; break;
  }
  merged[((size_t)b * LVOX + l) * (NSCAN * NC) + blk * NC + c] =
      hid[(((size_t)b * NSCAN + s) * LVOX + lsrc) * NC + c];
}

// -------------------- BN + transpose to NCDHW + channel mean ---------------
__global__ __launch_bounds__(256) void k_bn_mean(
    const float* __restrict__ mres, const float* __restrict__ w,
    const float* __restrict__ bb, const float* __restrict__ mean,
    const float* __restrict__ var, float* __restrict__ outp,
    float* __restrict__ means) {
  int bc = blockIdx.x;               // b*96 + c
  int c = bc % NC;
  int b = bc / NC;
  float sc = rsqrtf(var[c] + EPSF) * w[c];
  float mn = mean[c], bv = bb[c];
  float local = 0.f;
  for (int l = threadIdx.x; l < LVOX; l += 256) {
    float v = (mres[((size_t)b * LVOX + l) * NC + c] - mn) * sc + bv;
    outp[(size_t)bc * LVOX + l] = v;
    local += v;
  }
  __shared__ float red[256];
  red[threadIdx.x] = local;
  __syncthreads();
  for (int off = 128; off > 0; off >>= 1) {
    if (threadIdx.x < off) red[threadIdx.x] += red[threadIdx.x + off];
    __syncthreads();
  }
  if (threadIdx.x == 0) means[bc] = red[0] * (1.f / LVOX);
}

// ------------------------------ squeeze-excite -----------------------------
__global__ void k_se(const float* __restrict__ means, const float* __restrict__ w1,
                     const float* __restrict__ w2, float* __restrict__ scales) {
  __shared__ float sm[BB * NC];
  __shared__ float sh[BB * 6];
  int tid = threadIdx.x;
  if (tid < BB * NC) sm[tid] = means[tid];
  __syncthreads();
  if (tid < BB * 6) {
    int b = tid / 6, jj = tid % 6;
    float acc = 0.f;
    for (int c = 0; c < NC; ++c) acc += sm[b * NC + c] * w1[jj * NC + c];
    sh[tid] = (acc > 0.f) ? acc : 0.01f * acc;   // leaky relu
  }
  __syncthreads();
  if (tid < BB * NC) {
    int b = tid / NC, c = tid % NC;
    float acc = 0.f;
    #pragma unroll
    for (int jj = 0; jj < 6; ++jj) acc += sh[b * 6 + jj] * w2[c * 6 + jj];
    scales[tid] = 1.f / (1.f + expf(-acc));
  }
}

__global__ void k_scale(float* __restrict__ outp, const float* __restrict__ scales) {
  int idx = blockIdx.x * blockDim.x + threadIdx.x;
  if (idx >= BB * NC * LVOX) return;
  outp[idx] *= scales[idx / LVOX];
}

// ------------------------------- launcher ----------------------------------
extern "C" void kernel_launch(void* const* d_in, const int* in_sizes, int n_in,
                              void* d_out, int out_size, void* d_ws, size_t ws_size,
                              hipStream_t stream) {
  const float* x       = (const float*)d_in[0];
  const float* ds_w    = (const float*)d_in[1];
  const float* ds_b    = (const float*)d_in[2];
  const float* ds_bnw  = (const float*)d_in[3];
  const float* ds_bnb  = (const float*)d_in[4];
  const float* ds_bnm  = (const float*)d_in[5];
  const float* ds_bnv  = (const float*)d_in[6];
  const float* ln1w    = (const float*)d_in[7];
  const float* ln1b    = (const float*)d_in[8];
  const float* inpw    = (const float*)d_in[9];
  const float* convw   = (const float*)d_in[10];
  const float* convb   = (const float*)d_in[11];
  const float* dtbias  = (const float*)d_in[12];
  const float* alog    = (const float*)d_in[13];
  const float* Dparam  = (const float*)d_in[14];
  const float* mnw     = (const float*)d_in[15];
  const float* outpw   = (const float*)d_in[16];
  const float* ln2w    = (const float*)d_in[17];
  const float* ln2b    = (const float*)d_in[18];
  const float* fc1w    = (const float*)d_in[19];
  const float* fc1b    = (const float*)d_in[20];
  const float* fc2w    = (const float*)d_in[21];
  const float* fc2b    = (const float*)d_in[22];
  const float* normfw  = (const float*)d_in[23];
  const float* normfb  = (const float*)d_in[24];
  const float* mcw     = (const float*)d_in[25];
  const float* mcb     = (const float*)d_in[26];
  const float* mbnw    = (const float*)d_in[27];
  const float* mbnb    = (const float*)d_in[28];
  const float* mbnm    = (const float*)d_in[29];
  const float* mbnv    = (const float*)d_in[30];
  const float* sew1    = (const float*)d_in[31];
  const float* sew2    = (const float*)d_in[32];
  float* out = (float*)d_out;

  // workspace arena (floats); total ~64M floats (~255 MB)
  float* ws = (float*)d_ws;
  size_t off = 0;
  auto alloc = [&](size_t n) { float* p = ws + off; off += n; return p; };
  float* xin    = alloc((size_t)BB * NC * LVOX);
  float* hiddn  = alloc((size_t)MROWS * NC);            // seq / hidden
  float* resid  = alloc((size_t)MROWS * NC);
  float* normb  = alloc((size_t)MROWS * NC);
  float* zxb    = alloc((size_t)MROWS * DINPROJ);
  float* xbcc   = alloc((size_t)MROWS * CONVDIM);
  float* dtbuf  = alloc((size_t)MROWS * NHEADS);
  float* dabuf  = alloc((size_t)MROWS * NHEADS);
  float* ysbuf  = alloc((size_t)MROWS * DINNER);        // also MLP hidden
  float* ynb    = alloc((size_t)MROWS * DINNER);
  float* meansb = alloc(BB * NC);
  float* scaleb = alloc(BB * NC);
  float* merged = zxb;    // reuse: 3456*1152 <= MROWS*520
  float* mres   = xbcc;   // reuse: 3456*96  <= MROWS*320

  dim3 blk(256);
  int nvox = BB * NC * LVOX;

  k_dsconv_bn<<<(nvox + 255) / 256, blk, 0, stream>>>(
      x, ds_w, ds_b, ds_bnw, ds_bnb, ds_bnm, ds_bnv, xin);
  k_cross_scan<<<(nvox + 255) / 256, blk, 0, stream>>>(xin, hiddn);

  for (int d = 0; d < 2; ++d) {
    k_ln_residual<<<MROWS / 8, blk, 0, stream>>>(
        hiddn, (d == 0) ? nullptr : resid, resid, ln1w + d * NC, ln1b + d * NC, normb);
    k_gemm_wmma<NC><<<dim3(MROWS / BM, (DINPROJ + BN - 1) / BN), blk, 0, stream>>>(
        normb, inpw + (size_t)d * DINPROJ * NC, nullptr, zxb, DINPROJ, 0);
    k_conv1d_silu<<<(BB * TSEQ * CONVDIM + 255) / 256, blk, 0, stream>>>(
        zxb, convw + d * CONVDIM * DCONV, convb + d * CONVDIM, xbcc);
    k_dt<<<(MROWS * NHEADS + 255) / 256, blk, 0, stream>>>(
        zxb, dtbias + d * NHEADS, alog + d * NHEADS, dtbuf, dabuf);
    k_scan<<<BB * NHEADS, blk, 0, stream>>>(xbcc, dtbuf, dabuf, ysbuf);
    k_gate_rms<<<MROWS / 8, blk, 0, stream>>>(
        ysbuf, xbcc, zxb, Dparam + d * NHEADS, mnw + d * DINNER, ynb);
    k_gemm_wmma<DINNER><<<dim3(MROWS / BM, (NC + BN - 1) / BN), blk, 0, stream>>>(
        ynb, outpw + (size_t)d * NC * DINNER, nullptr, hiddn, NC, 0);
    k_ln_residual<<<MROWS / 8, blk, 0, stream>>>(
        hiddn, resid, resid, ln2w + d * NC, ln2b + d * NC, normb);
    k_gemm_wmma<NC><<<dim3(MROWS / BM, (DINNER + BN - 1) / BN), blk, 0, stream>>>(
        normb, fc1w + (size_t)d * DINNER * NC, fc1b + d * DINNER, ysbuf,
        DINNER, /*gelu*/1);
    k_gemm_wmma<DINNER><<<dim3(MROWS / BM, (NC + BN - 1) / BN), blk, 0, stream>>>(
        ysbuf, fc2w + (size_t)d * NC * DINNER, fc2b + d * NC, hiddn, NC, 0);
  }

  k_ln_residual<<<MROWS / 8, blk, 0, stream>>>(
      hiddn, resid, resid, normfw, normfb, normb);
  k_cross_merge<<<(BB * NSCAN * LVOX * NC + 255) / 256, blk, 0, stream>>>(normb, merged);
  k_gemm_wmma<NSCAN * NC><<<dim3(BB * LVOX / BM, (NC + BN - 1) / BN), blk, 0, stream>>>(
      merged, mcw, mcb, mres, NC, 0);
  k_bn_mean<<<BB * NC, blk, 0, stream>>>(mres, mbnw, mbnb, mbnm, mbnv, out, meansb);
  k_se<<<1, blk, 0, stream>>>(meansb, sew1, sew2, scaleb);
  k_scale<<<(nvox + 255) / 256, blk, 0, stream>>>(out, scaleb);
}
